// AttentionBlock_88330297409616
// MI455X (gfx1250) — matrix operand order repaired and checked by reading, not compiled, not run on hardware
//
#include <hip/hip_runtime.h>

#define SEQ 2048
#define DM  1024
#define HID 1024

typedef __attribute__((ext_vector_type(16))) __bf16 v16bf;
typedef __attribute__((ext_vector_type(8)))  __bf16 v8bf;
typedef __attribute__((ext_vector_type(4)))  __bf16 v4bf;
typedef __attribute__((ext_vector_type(8)))  float  v8f;

typedef unsigned int u32x4 __attribute__((ext_vector_type(4)));
typedef int          i32x4 __attribute__((ext_vector_type(4)));
typedef int          i32x8 __attribute__((ext_vector_type(8)));

#define LDA 40   // padded LDS row stride (bf16 elems): 80 B rows, 16B-aligned, conflict-free

// ---------------------------------------------------------------------------
// TDM: async-load one 128x32 bf16 tile (row stride = rowStrideElems) from
// global memory into LDS at byte offset ldsOff, inserting 16B of padding after
// every 64B row so the LDS layout has an 80B row stride (matches LDA=40).
// D# layout per CDNA5 ISA ch.8. Issue from ONE wave only; tracked by TENSORcnt.
// ---------------------------------------------------------------------------
__device__ __forceinline__ void tdm_tile_load_128x32(const __bf16* gsrc,
                                                     unsigned ldsOff,
                                                     unsigned rowStrideElems)
{
  const unsigned long long ga = (unsigned long long)(uintptr_t)gsrc;

  u32x4 g0;
  g0[0] = 1u;                                   // count=1, user descriptor
  g0[1] = ldsOff;                               // lds_addr [63:32]
  g0[2] = (unsigned)ga;                         // global_addr lo
  g0[3] = (unsigned)((ga >> 32) & 0x01FFFFFFu)  // global_addr [56:32]
        | (2u << 30);                           // type=2 ("image")

  i32x8 g1;
  g1[0] = (int)((1u << 16)      // data_size = 1 -> 2 bytes
              | (1u << 20)      // pad_enable
              | (3u << 22)      // pad_interval = 3 -> every 16 DWORDs (64 B)
              | (3u << 25));    // pad_amount   = 3 -> 4 DWORDs (16 B)
  g1[1] = (int)(32u << 16);     // tensor_dim0 = 32   (bits 63:48 -> lo16)
  g1[2] = (int)(128u << 16);    // tensor_dim0 hi = 0; tensor_dim1 = 128 (lo16)
  g1[3] = (int)(32u << 16);     // tensor_dim1 hi = 0; tile_dim0 = 32
  g1[4] = 128;                  // tile_dim1 = 128; tile_dim2 = 0
  g1[5] = (int)rowStrideElems;  // tensor_dim0_stride lo32 (elements)
  g1[6] = 0;                    // stride hi / tensor_dim1_stride lo
  g1[7] = 0;

  const i32x4 z4 = {0, 0, 0, 0};
#if __has_include(<hip/amd_detail/amd_gfx1250_TDM.h>)
  const i32x8 z8 = {0, 0, 0, 0, 0, 0, 0, 0};
  __builtin_amdgcn_tensor_load_to_lds(g0, g1, z4, z4, z8, 0);   // clang-23 arity
#else
  __builtin_amdgcn_tensor_load_to_lds(g0, g1, z4, z4, 0);       // ROCm 7.2 arity
#endif
}

// ---------------------------------------------------------------------------
// Generic D = A * B^T (+bias) GEMM on the bf16 WMMA pipe, double-buffered LDS.
//  A: [M,K] row-major (f32 or bf16), B: [N,K] row-major (f32 or bf16)
//  f32 operands: staged by all threads (clause of b128 loads, then cvt+ds_store).
//  bf16 operands: staged by the Tensor Data Mover (async, TENSORcnt).
//  C: f32 [M,N], bf16 [M,N], or bf16 transposed [N,M]
// ---------------------------------------------------------------------------
template<int A_F32, int B_F32, int BIAS, int OUT_BF16, int OUT_TRANS>
__global__ __launch_bounds__(256)
void gemm_wmma_bf16(const void* __restrict__ Ap, const void* __restrict__ Bp,
                    const float* __restrict__ bias, void* __restrict__ Cp,
                    int M, int N, int K,
                    long long sA, long long sB, long long sC)
{
  __shared__ __bf16 lA[2][128][LDA];
  __shared__ __bf16 lB[2][128][LDA];

  const int t  = threadIdx.x;
  const int bz = blockIdx.z;
  const int m0 = blockIdx.y * 128;
  const int n0 = blockIdx.x * 128;

  const float*  Af = (const float*)Ap  + (size_t)bz * sA;
  const __bf16* Ab = (const __bf16*)Ap + (size_t)bz * sA;
  const float*  Bf = (const float*)Bp  + (size_t)bz * sB;
  const __bf16* Bb = (const __bf16*)Bp + (size_t)bz * sB;

  const int r0 = t >> 3;          // 0..31: staging row base
  const int cg = (t & 7) * 4;     // staging col group (4 elems)

  const int lane = t & 31;
  const int lo   = lane & 15;
  const int hi   = lane >> 4;
  const int wave = t >> 5;        // 8 waves: 4 (M) x 2 (N)
  const int wm   = wave & 3;
  const int wn   = wave >> 2;

  constexpr int NTDM = (A_F32 ? 0 : 1) + (B_F32 ? 0 : 1);

  // Stage one 128x32 K-slab of A and B into LDS buffer `buf`.
  // Phase 1 gathers ALL global loads into registers (one clause, one wait),
  // phase 2 converts and writes LDS. Keeps load latency off the critical path.
  auto stage = [&](int buf, int kk) {
    float4 xa[4], xb[4];
    if constexpr (A_F32) {
      #pragma unroll
      for (int rr = 0; rr < 4; ++rr)
        xa[rr] = *(const float4*)(Af + (size_t)(m0 + r0 + rr * 32) * K + kk + cg);
    }
    if constexpr (B_F32) {
      #pragma unroll
      for (int rr = 0; rr < 4; ++rr)
        xb[rr] = *(const float4*)(Bf + (size_t)(n0 + r0 + rr * 32) * K + kk + cg);
    }
    if constexpr (A_F32) {
      #pragma unroll
      for (int rr = 0; rr < 4; ++rr) {
        v4bf va;
        va[0] = (__bf16)xa[rr].x; va[1] = (__bf16)xa[rr].y;
        va[2] = (__bf16)xa[rr].z; va[3] = (__bf16)xa[rr].w;
        *(v4bf*)&lA[buf][r0 + rr * 32][cg] = va;
      }
    }
    if constexpr (B_F32) {
      #pragma unroll
      for (int rr = 0; rr < 4; ++rr) {
        v4bf vb;
        vb[0] = (__bf16)xb[rr].x; vb[1] = (__bf16)xb[rr].y;
        vb[2] = (__bf16)xb[rr].z; vb[3] = (__bf16)xb[rr].w;
        *(v4bf*)&lB[buf][r0 + rr * 32][cg] = vb;
      }
    }
    if constexpr (NTDM > 0) {
      if (t < 32) {   // one wave issues the TDM ops (EXEC-independent op: keep behind a branch)
        if constexpr (!A_F32)
          tdm_tile_load_128x32(Ab + (size_t)m0 * K + kk,
                               (unsigned)(uintptr_t)&lA[buf][0][0], (unsigned)K);
        if constexpr (!B_F32)
          tdm_tile_load_128x32(Bb + (size_t)n0 * K + kk,
                               (unsigned)(uintptr_t)&lB[buf][0][0], (unsigned)K);
      }
    }
  };

  v8f acc[2][4] = {};

  stage(0, 0);                                  // prologue: fill buffer 0
  for (int kk = 0; kk < K; kk += 32) {
    const int  cur  = (kk >> 5) & 1;
    const bool more = (kk + 32) < K;
    if (more) stage(cur ^ 1, kk + 32);          // prefetch next slab into other buffer

    if constexpr (NTDM > 0) {
      if (t < 32) {                             // issuing wave: current slab's TDM ops done
        if (more) __builtin_amdgcn_s_wait_tensorcnt(NTDM);
        else      __builtin_amdgcn_s_wait_tensorcnt(0);
      }
    }
    __syncthreads();                            // cur slab visible to all waves

    // ---- A fragments (16x32 bf16): lanes 0-15 k{0..7,16..23}, lanes 16-31 k{8..15,24..31}
    v16bf afr[2];
    #pragma unroll
    for (int mt = 0; mt < 2; ++mt) {
      const int row = wm * 32 + mt * 16 + lo;
      const v8bf a0 = *(const v8bf*)&lA[cur][row][hi * 8];
      const v8bf a1 = *(const v8bf*)&lA[cur][row][16 + hi * 8];
      #pragma unroll
      for (int i = 0; i < 8; ++i) { afr[mt][i] = a0[i]; afr[mt][i + 8] = a1[i]; }
    }
    // ---- B fragments (32x16 bf16): lane = column n; lanes 0-15 k{0..7,16..23}, lanes 16-31 k{8..15,24..31}
    v16bf bfr[4];
    #pragma unroll
    for (int nt = 0; nt < 4; ++nt) {
      const int row = wn * 64 + nt * 16 + lo;
      const v8bf b0 = *(const v8bf*)&lB[cur][row][hi * 8];
      const v8bf b1 = *(const v8bf*)&lB[cur][row][16 + hi * 8];
      #pragma unroll
      for (int i = 0; i < 8; ++i) { bfr[nt][i] = b0[i]; bfr[nt][i + 8] = b1[i]; }
    }

    #pragma unroll
    for (int mt = 0; mt < 2; ++mt)
      #pragma unroll
      for (int nt = 0; nt < 4; ++nt)
        acc[mt][nt] = __builtin_amdgcn_wmma_f32_16x16x32_bf16(
            false, afr[mt], false, bfr[nt], (short)0, acc[mt][nt], false, false);

    __syncthreads();                            // reads done before buffer is overwritten
  }

  // ---- epilogue: C layout per lane: n = lo, m = hi*8 + r (r = accum element)
  float*  Cf = (float*)Cp  + (size_t)bz * sC;
  __bf16* Cb = (__bf16*)Cp + (size_t)bz * sC;

  #pragma unroll
  for (int mt = 0; mt < 2; ++mt) {
    const int mb = m0 + wm * 32 + mt * 16 + hi * 8;
    #pragma unroll
    for (int nt = 0; nt < 4; ++nt) {
      const int n = n0 + wn * 64 + nt * 16 + lo;
      float bv = 0.0f;
      if constexpr (BIAS) bv = bias[n];
      if constexpr (OUT_BF16 && OUT_TRANS) {
        // transposed: each lane owns 8 consecutive m for fixed n -> one packed store
        v8bf o;
        #pragma unroll
        for (int r = 0; r < 8; ++r) o[r] = (__bf16)(acc[mt][nt][r] + bv);
        *(v8bf*)&Cb[(size_t)n * M + mb] = o;
      } else if constexpr (OUT_BF16) {
        #pragma unroll
        for (int r = 0; r < 8; ++r)
          Cb[(size_t)(mb + r) * N + n] = (__bf16)(acc[mt][nt][r] + bv);
      } else {
        #pragma unroll
        for (int r = 0; r < 8; ++r)
          Cf[(size_t)(mb + r) * N + n] = acc[mt][nt][r] + bv;
      }
    }
  }
}

// softmax over axis=1 (query axis): column softmax of dot[b, :, k] / 32, in place.
__global__ __launch_bounds__(256)
void softmax_axis1(float* __restrict__ w, int S, float inv_scale)
{
  const int b = blockIdx.y;
  const int k = blockIdx.x * blockDim.x + threadIdx.x;
  float* col = w + (size_t)b * S * S + k;
  float m = -3.4e38f, s = 0.0f;
  for (int q = 0; q < S; ++q) {
    const float x  = col[(size_t)q * S] * inv_scale;
    const float mn = fmaxf(m, x);
    s = s * __expf(m - mn) + __expf(x - mn);
    m = mn;
  }
  const float inv = 1.0f / s;
  for (int q = 0; q < S; ++q) {
    const float x = col[(size_t)q * S] * inv_scale;
    col[(size_t)q * S] = __expf(x - m) * inv;
  }
}

extern "C" void kernel_launch(void* const* d_in, const int* in_sizes, int n_in,
                              void* d_out, int out_size, void* d_ws, size_t ws_size,
                              hipStream_t stream)
{
  const float* q  = (const float*)d_in[0];
  const float* k  = (const float*)d_in[1];
  const float* v  = (const float*)d_in[2];
  const float* Wq = (const float*)d_in[3];
  const float* bq = (const float*)d_in[4];
  const float* Wk = (const float*)d_in[5];
  const float* bk = (const float*)d_in[6];
  const float* Wv = (const float*)d_in[7];
  const float* bv = (const float*)d_in[8];

  const int S = SEQ, D = DM, H = HID;
  const int B = in_sizes[0] / (S * D);

  float* attn_out = (float*)d_out;                      // [B,S,H]
  float* attn_w   = (float*)d_out + (size_t)B * S * H;  // [B,S,S]

  __bf16* Qb = (__bf16*)d_ws;                           // [B,S,H] bf16
  __bf16* Kb = Qb + (size_t)B * S * H;                  // [B,S,H] bf16
  __bf16* Vt = Kb + (size_t)B * S * H;                  // [B,H,S] bf16 (transposed)

  const dim3 blk(256);

  // 1) projections: Q = qWq^T+b, K = kWk^T+b  (bf16), V stored transposed
  const dim3 gProj(H / 128, S / 128, B);
  gemm_wmma_bf16<1,1,1,1,0><<<gProj, blk, 0, stream>>>(q, Wq, bq, Qb, S, H, D,
      (long long)S * D, 0LL, (long long)S * H);
  gemm_wmma_bf16<1,1,1,1,0><<<gProj, blk, 0, stream>>>(k, Wk, bk, Kb, S, H, D,
      (long long)S * D, 0LL, (long long)S * H);
  gemm_wmma_bf16<1,1,1,1,1><<<gProj, blk, 0, stream>>>(v, Wv, bv, Vt, S, H, D,
      (long long)S * D, 0LL, (long long)H * S);

  // 2) scores: dot = Q K^T  -> f32 into attn_weights region (A and B via TDM)
  const dim3 gDot(S / 128, S / 128, B);
  gemm_wmma_bf16<0,0,0,0,0><<<gDot, blk, 0, stream>>>(Qb, Kb, nullptr, attn_w, S, S, H,
      (long long)S * H, (long long)S * H, (long long)S * S);

  // 3) softmax over query axis, scale = 1/sqrt(1024) = 1/32, in place
  const dim3 gSm(S / 256, B);
  softmax_axis1<<<gSm, blk, 0, stream>>>(attn_w, S, 1.0f / 32.0f);

  // 4) attn_out = attn_w * Vt^T  (A f32 -> bf16 on the fly, B via TDM)
  const dim3 gOut(H / 128, S / 128, B);
  gemm_wmma_bf16<1,0,0,0,0><<<gOut, blk, 0, stream>>>(attn_w, Vt, nullptr, attn_out, S, H, S,
      (long long)S * S, (long long)H * S, (long long)S * H);
}
